// Generator_20435454394819
// MI455X (gfx1250) — matrix operand order repaired and checked by reading, not compiled, not run on hardware
//
#include <hip/hip_runtime.h>
#include <math.h>

// ---------------------------------------------------------------------------
// StyleGAN2 modulated upsample-conv block, fused for MI455X (gfx1250).
//
// convT(stride2) + 4x4 blur  ==  four parity-specific 3x3 convs on x:
//   y[2*by+py, 2*bx+px] = sum_{cin,ty,tx} Weff[py][px][cout][cin][ty][tx]
//                         * x[cin][by+ty-1][bx+tx-1] * s[b][cin]
// Weff = (flipped 3x3 weight) (*) (flipped 4x4 blur), parity-sliced; demod is
// applied post-GEMM (linear => exact). Main loop: TDM stages A (weights) and
// B (activations) tiles into double-buffered LDS; 8 waves run 4 parity WMMAs
// per K-chunk; epilogue stores coalesced float2 (both x-parities).
// ---------------------------------------------------------------------------

#define B_      8
#define CIN     512
#define COUT    512
#define H_      64
#define W_      64
#define SDIM    512
#define HOUT    128
#define WOUT    128
#define KTOT    (CIN * 9)        // 4608, tap-major: k = tap*512 + cin
#define NKB     (KTOT / 32)      // 144 K-chunks of 32
#define NMB     (COUT / 16)      // 32 M-blocks of 16
#define XP      66               // padded spatial extent (pad ring of 1)

#define CONV_SCALE  0.014731391274719739f   // 1/sqrt(512*9)
#define CONV_SCALE2 2.170138888888889e-4f   // 1/4608
#define INV_SQRT_S  0.04419417382415922f    // 1/sqrt(512)
#define LRELU_S     0.2f
#define GAIN_S      1.4142135623730951f

typedef __attribute__((ext_vector_type(16))) __bf16 v16bf;
typedef __attribute__((ext_vector_type(8)))  float  v8f;
typedef unsigned int u32x4 __attribute__((ext_vector_type(4)));
typedef int          i32x4 __attribute__((ext_vector_type(4)));
typedef int          i32x8 __attribute__((ext_vector_type(8)));

#if defined(__has_builtin)
#if __has_builtin(__builtin_amdgcn_tensor_load_to_lds)
#define USE_TDM 1
#endif
#endif
#ifndef USE_TDM
#define USE_TDM 0
#endif

static __device__ __forceinline__ __bf16 f2bf(float f) {
  union { float f; unsigned u; } v; v.f = f;
  unsigned r = v.u + 0x7FFFu + ((v.u >> 16) & 1u);   // round-to-nearest-even
  union { unsigned short s; __bf16 b; } o; o.s = (unsigned short)(r >> 16);
  return o.b;
}

// --------------------------- prep kernels ----------------------------------

__global__ void style_kernel(const float* __restrict__ style,
                             const float* __restrict__ mod_weight,
                             const float* __restrict__ mod_bias,
                             float* __restrict__ s_out) {
  __shared__ float sty[SDIM];
  const int b = blockIdx.x;
  const int cin = threadIdx.x;            // blockDim.x == 512
  sty[cin] = style[b * SDIM + cin];
  __syncthreads();
  const float* mw = mod_weight + (size_t)cin * SDIM;
  float acc = 0.f;
  #pragma unroll 8
  for (int t = 0; t < SDIM; ++t) acc += sty[t] * mw[t];
  s_out[b * CIN + cin] = acc * INV_SQRT_S + mod_bias[cin];
}

__global__ void wsq_kernel(const float* __restrict__ weight,
                           float* __restrict__ wsq) {
  const int cout = blockIdx.x, cin = threadIdx.x;   // 512 x 512
  const float* w = weight + ((size_t)cout * CIN + cin) * 9;
  float a = 0.f;
  #pragma unroll
  for (int t = 0; t < 9; ++t) { float v = w[t]; a += v * v; }
  wsq[(size_t)cout * CIN + cin] = a * CONV_SCALE2;
}

__global__ void demod_kernel(const float* __restrict__ wsq,
                             const float* __restrict__ s,
                             float* __restrict__ demod) {
  __shared__ float s2[CIN];
  const int b = blockIdx.x;
  const int cout = blockIdx.y * 128 + threadIdx.x;  // grid (B,4) x 128
  for (int i = threadIdx.x; i < CIN; i += 128) {
    float v = s[b * CIN + i]; s2[i] = v * v;
  }
  __syncthreads();
  const float* wr = wsq + (size_t)cout * CIN;
  float a = 0.f;
  #pragma unroll 8
  for (int i = 0; i < CIN; ++i) a += wr[i] * s2[i];
  demod[b * COUT + cout] = rsqrtf(a + 1e-8f);
}

// Modulated input, NHWC bf16 with zero pad ring: xs_t[b][y][x][cin], y,x<66
__global__ void pack_xs_kernel(const float* __restrict__ x,
                               const float* __restrict__ s,
                               __bf16* __restrict__ xs_t) {
  const int y = blockIdx.x;               // 0..65
  const int b = blockIdx.y;
  const int iy = y - 1;
  const bool yok = (iy >= 0) & (iy < H_);
  for (int cin = threadIdx.x; cin < CIN; cin += 256) {
    const float sv = s[b * CIN + cin];
    const float* xr = x + (((size_t)b * CIN + cin) * H_ + (yok ? iy : 0)) * W_;
    __bf16* dst = xs_t + ((size_t)(b * XP + y) * XP) * CIN + cin;
    for (int xp = 0; xp < XP; ++xp) {
      const int ix = xp - 1;
      float v = (yok && ix >= 0 && ix < W_) ? xr[ix] * sv : 0.f;
      dst[(size_t)xp * CIN] = f2bf(v);
    }
  }
}

// Effective parity kernels pre-swizzled into WMMA A-fragment lane order.
// apack[p][mb][kb][lane][e]; K = tap*512 + cin.
__global__ void pack_weff_kernel(const float* __restrict__ weight,
                                 __bf16* __restrict__ apack) {
  const int kb = blockIdx.x, mb = blockIdx.y, p = blockIdx.z;  // 144,32,4
  const int lane = threadIdx.x;                                // 32
  const int py = p >> 1, px = p & 1;
  const int cout = mb * 16 + (lane & 15);
  const int grp = lane >> 4;
  const float b1[4] = {0.25f, 0.75f, 0.75f, 0.25f};  // [1,3,3,1]/4 (symmetric)
  v16bf outv;
  #pragma unroll
  for (int e = 0; e < 16; ++e) {
    const int v = e >> 1, lo = e & 1;
    const int kl = 16 * grp + e;
    const int k = kb * 32 + kl;
    const int tap = k >> 9;          // 0..8
    const int cin = k & 511;
    const int ty = tap / 3, tx = tap % 3;
    const int my = 2 * ty + 1 - py;
    const int mx = 2 * tx + 1 - px;
    const float* w = weight + ((size_t)cout * CIN + cin) * 9;
    float acc = 0.f;
    #pragma unroll
    for (int ky = 0; ky < 3; ++ky) {
      const int jy = my - ky;
      if (jy < 0 || jy > 3) continue;
      #pragma unroll
      for (int kx = 0; kx < 3; ++kx) {
        const int jx = mx - kx;
        if (jx < 0 || jx > 3) continue;
        acc += w[(2 - ky) * 3 + (2 - kx)] * b1[jy] * b1[jx];
      }
    }
    outv[e] = f2bf(acc * CONV_SCALE);
  }
  v16bf* dst = reinterpret_cast<v16bf*>(apack) +
               (((size_t)(p * NMB + mb) * NKB + kb) * 32 + lane);
  *dst = outv;
}

// --------------------------- TDM helper ------------------------------------

#if USE_TDM
// Build a D# (ISA ch.8) and issue TENSOR_LOAD_TO_LDS. Up-to-3D tile; strides
// and dims in data_size units; dsz_code: 0=1B,1=2B,2=4B,3=8B.
static __device__ __forceinline__ void tdm_load_3d(
    unsigned lds_addr, unsigned long long ga,
    unsigned dim0, unsigned dim1, unsigned dim2,
    unsigned tile0, unsigned tile1, unsigned tile2,
    unsigned long long stride0, unsigned long long stride1,
    unsigned dsz_code) {
  u32x4 g0;
  g0[0] = 1u;                                        // count=1, user mode
  g0[1] = lds_addr;                                  // LDS byte address
  g0[2] = (unsigned)(ga & 0xFFFFFFFFull);            // global_addr[31:0]
  g0[3] = (unsigned)((ga >> 32) & 0x01FFFFFFull) | 0x80000000u; // [56:32],type=2
  i32x8 g1;
  g1[0] = (int)(dsz_code << 16);                     // wg_mask=0, no pad/iter
  g1[1] = (int)((dim0 & 0xFFFFu) << 16);             // tensor_dim0 lo
  g1[2] = (int)(((dim0 >> 16) & 0xFFFFu) | ((dim1 & 0xFFFFu) << 16));
  g1[3] = (int)(((dim1 >> 16) & 0xFFFFu) | (tile0 << 16));
  g1[4] = (int)((tile1 & 0xFFFFu) | (tile2 << 16));
  g1[5] = (int)(stride0 & 0xFFFFFFFFull);            // tensor_dim0_stride
  g1[6] = (int)(((stride0 >> 32) & 0xFFFFull) | ((stride1 & 0xFFFFull) << 16));
  g1[7] = (int)((stride1 >> 16) & 0xFFFFFFFFull);    // tensor_dim1_stride hi
  i32x4 g2;
  g2[0] = (int)dim2;                                 // tensor_dim2
  g2[1] = 0; g2[2] = 0; g2[3] = 0;                   // no dim3, tile_dim3=0
  i32x4 g3 = {0, 0, 0, 0};
#if __has_include(<hip/amd_detail/amd_gfx1250_TDM.h>)
  i32x8 g4 = {0, 0, 0, 0, 0, 0, 0, 0};               // therock toolchain: 6 args
  __builtin_amdgcn_tensor_load_to_lds(g0, g1, g2, g3, g4, 0);
#else
  __builtin_amdgcn_tensor_load_to_lds(g0, g1, g2, g3, 0);  // ROCm 7.2: 5 args
#endif
}
#endif

// --------------------------- main WMMA kernel ------------------------------

// Block: 256 threads = 8 waves; wave -> 16 couts; block -> 128 couts x
// 16 spatial bases (row of 16). 4 parity accumulators share one B fragment.
__global__ __launch_bounds__(256) void modconv_wmma_kernel(
    const __bf16* __restrict__ xs_t, const __bf16* __restrict__ apack,
    const float* __restrict__ demod, const float* __restrict__ noise,
    const float* __restrict__ noise_w, const float* __restrict__ act_bias,
    float* __restrict__ out) {
  const int b = blockIdx.z;
  const int by = blockIdx.x >> 2;              // base row 0..63
  const int xt = blockIdx.x & 3;               // row tile of 16
  const int wave = threadIdx.x >> 5;
  const int lane = threadIdx.x & 31;
  const int mb0 = blockIdx.y * 8;
  const int mb = mb0 + wave;                   // M block (16 couts)
  const int n = lane & 15;                     // N index within tile
  const int grp = lane >> 4;
  const int bx = xt * 16 + n;

  v8f acc0 = {}, acc1 = {}, acc2 = {}, acc3 = {};
  const size_t xs_b = (size_t)b * XP * XP * CIN;   // element offset of image

#if USE_TDM
  // Double-buffered LDS staging, filled by the Tensor Data Mover.
  __shared__ __align__(1024) unsigned char lds_a[2][4][8][1024];  // [p][mb][..]
  __shared__ __align__(1024) unsigned char lds_b[2][16][1024];    // [n][cin*2]
  const unsigned long long a_ga0 = (unsigned long long)(uintptr_t)apack;
  const unsigned long long b_ga0 =
      (unsigned long long)(uintptr_t)xs_t + (unsigned long long)xs_b * 2ull;

  // A: 3D tile {x:512 elems}x{y:8 mb, stride NKB*512}x{z:4 p, stride 32*NKB*512}
  // B: linear 16 KB (16 consecutive x-positions, all 512 cins) for one tap
  if (wave == 0) {
    tdm_load_3d((unsigned)(uintptr_t)&lds_a[0][0][0][0],
                a_ga0 + (unsigned long long)(mb0 * NKB) * 1024ull,
                512u, 8u, 4u, 512u, 8u, 4u,
                (unsigned long long)NKB * 512ull,
                (unsigned long long)NMB * NKB * 512ull, 1u);
    tdm_load_3d((unsigned)(uintptr_t)&lds_b[0][0][0],
                b_ga0 + (unsigned long long)(by * XP + xt * 16) * 1024ull,
                8192u, 0u, 0u, 8192u, 0u, 0u, 0ull, 0ull, 1u);
  }
  int abuf = 0, bbuf = 0;
#endif

  for (int kb = 0; kb < NKB; ++kb) {
#if USE_TDM
    __builtin_amdgcn_s_wait_tensorcnt(0);     // wave0's in-flight TDMs done
    __syncthreads();                          // publish staged LDS buffers
    if (wave == 0 && kb + 1 < NKB) {
      tdm_load_3d((unsigned)(uintptr_t)&lds_a[abuf ^ 1][0][0][0],
                  a_ga0 + (unsigned long long)(mb0 * NKB + kb + 1) * 1024ull,
                  512u, 8u, 4u, 512u, 8u, 4u,
                  (unsigned long long)NKB * 512ull,
                  (unsigned long long)NMB * NKB * 512ull, 1u);
      if (((kb + 1) & 15) == 0) {             // next tap's activations
        const int tap1 = (kb + 1) >> 4;
        const int ty1 = tap1 / 3, tx1 = tap1 % 3;
        tdm_load_3d((unsigned)(uintptr_t)&lds_b[bbuf ^ 1][0][0],
                    b_ga0 + (unsigned long long)((by + ty1) * XP + xt * 16 + tx1) * 1024ull,
                    8192u, 0u, 0u, 8192u, 0u, 0u, 0ull, 0ull, 1u);
      }
    }
    const int koff = (kb & 15) * 32;          // cin offset within tap
    const v16bf bfrag = *reinterpret_cast<const v16bf*>(
        &lds_b[bbuf][n][(koff + grp * 16) * 2]);
    const v16bf a0 = *reinterpret_cast<const v16bf*>(&lds_a[abuf][0][wave][lane * 32]);
    const v16bf a1 = *reinterpret_cast<const v16bf*>(&lds_a[abuf][1][wave][lane * 32]);
    const v16bf a2 = *reinterpret_cast<const v16bf*>(&lds_a[abuf][2][wave][lane * 32]);
    const v16bf a3 = *reinterpret_cast<const v16bf*>(&lds_a[abuf][3][wave][lane * 32]);
    if (((kb + 1) & 15) == 0) bbuf ^= 1;
    abuf ^= 1;
#else
    const int kc = kb * 32;
    const int tap = kc >> 9;
    const int ty = tap / 3, tx = tap % 3;
    const int cin0 = (kc & 511) + grp * 16;
    const v16bf bfrag = *reinterpret_cast<const v16bf*>(
        xs_t + xs_b + ((size_t)((by + ty) * XP + (bx + tx))) * CIN + cin0);
    const v16bf* abase = reinterpret_cast<const v16bf*>(apack);
    const v16bf a0 = abase[(((size_t)(0 * NMB + mb) * NKB + kb) * 32 + lane)];
    const v16bf a1 = abase[(((size_t)(1 * NMB + mb) * NKB + kb) * 32 + lane)];
    const v16bf a2 = abase[(((size_t)(2 * NMB + mb) * NKB + kb) * 32 + lane)];
    const v16bf a3 = abase[(((size_t)(3 * NMB + mb) * NKB + kb) * 32 + lane)];
#endif
    acc0 = __builtin_amdgcn_wmma_f32_16x16x32_bf16(false, a0, false, bfrag,
                                                   (short)0, acc0, false, false);
    acc1 = __builtin_amdgcn_wmma_f32_16x16x32_bf16(false, a1, false, bfrag,
                                                   (short)0, acc1, false, false);
    acc2 = __builtin_amdgcn_wmma_f32_16x16x32_bf16(false, a2, false, bfrag,
                                                   (short)0, acc2, false, false);
    acc3 = __builtin_amdgcn_wmma_f32_16x16x32_bf16(false, a3, false, bfrag,
                                                   (short)0, acc3, false, false);
  }

  // Epilogue. D 16x16 f32: VGPR r -> cout = mb*16 + grp*8 + r, N = lane&15.
  // Pair the two x-parities into one coalesced float2 store per register.
  const float nw = noise_w[0];
  const int coutb = mb * 16 + grp * 8;
  float dem[8], ab[8];
  #pragma unroll
  for (int r = 0; r < 8; ++r) {
    dem[r] = demod[b * COUT + coutb + r];
    ab[r]  = act_bias[coutb + r];
  }
  const int oxe = 2 * bx;
  #pragma unroll
  for (int py = 0; py < 2; ++py) {
    const int oy = 2 * by + py;
    const float nzL = nw * noise[((size_t)b * HOUT + oy) * WOUT + oxe];
    const float nzR = nw * noise[((size_t)b * HOUT + oy) * WOUT + oxe + 1];
    const v8f aL = py ? acc2 : acc0;
    const v8f aR = py ? acc3 : acc1;
    #pragma unroll
    for (int r = 0; r < 8; ++r) {
      float vl = aL[r] * dem[r] + nzL + ab[r];
      float vr = aR[r] * dem[r] + nzR + ab[r];
      vl = (vl >= 0.f ? vl : LRELU_S * vl) * GAIN_S;
      vr = (vr >= 0.f ? vr : LRELU_S * vr) * GAIN_S;
      float2 o; o.x = vl; o.y = vr;
      *reinterpret_cast<float2*>(
          &out[(((size_t)b * COUT + coutb + r) * HOUT + oy) * WOUT + oxe]) = o;
    }
  }
}

// --------------------------- launcher --------------------------------------

extern "C" void kernel_launch(void* const* d_in, const int* in_sizes, int n_in,
                              void* d_out, int out_size, void* d_ws, size_t ws_size,
                              hipStream_t stream) {
  const float* x        = (const float*)d_in[0];
  const float* style    = (const float*)d_in[1];
  const float* weight   = (const float*)d_in[2];
  const float* mod_w    = (const float*)d_in[3];
  const float* mod_b    = (const float*)d_in[4];
  const float* noise    = (const float*)d_in[5];
  const float* noise_w  = (const float*)d_in[6];
  const float* act_bias = (const float*)d_in[7];
  float* out = (float*)d_out;

  char* ws = (char*)d_ws;
  size_t off = 0;
  auto carve = [&](size_t bytes) -> char* {
    char* p = ws + off;
    off = (off + bytes + 255) & ~(size_t)255;
    return p;
  };
  float*  s_buf  = (float*)carve((size_t)B_ * CIN * 4);
  float*  wsq    = (float*)carve((size_t)COUT * CIN * 4);
  float*  demod  = (float*)carve((size_t)B_ * COUT * 4);
  __bf16* xs_t   = (__bf16*)carve((size_t)B_ * XP * XP * CIN * 2);      // ~34 MB
  __bf16* apack  = (__bf16*)carve((size_t)4 * NMB * NKB * 32 * 16 * 2); // ~18 MB
  (void)ws_size; (void)in_sizes; (void)n_in; (void)out_size;

  style_kernel<<<dim3(B_), dim3(SDIM), 0, stream>>>(style, mod_w, mod_b, s_buf);
  wsq_kernel<<<dim3(COUT), dim3(CIN), 0, stream>>>(weight, wsq);
  demod_kernel<<<dim3(B_, 4), dim3(128), 0, stream>>>(wsq, s_buf, demod);
  pack_xs_kernel<<<dim3(XP, B_), dim3(256), 0, stream>>>(x, s_buf, xs_t);
  pack_weff_kernel<<<dim3(NKB, NMB, 4), dim3(32), 0, stream>>>(weight, apack);
  modconv_wmma_kernel<<<dim3(H_ * 4, COUT / 128, B_), dim3(256), 0, stream>>>(
      xs_t, apack, demod, noise, noise_w, act_bias, out);
}
